// VisionMambaClassifier_10986526343944
// MI455X (gfx1250) — matrix operand order repaired and checked by reading, not compiled, not run on hardware
//
#include <hip/hip_runtime.h>
#include <hip/hip_bf16.h>

// ---------------------------------------------------------------------------
// Vision-Mamba classifier for MI455X (gfx1250, wave32, WMMA).
// All GEMM operands pre-converted to bf16 so fragment loads are raw B128s.
// conv2 implicit GEMM uses tap-major K ordering + channel-last h1 so its
// im2col A-fragments are single B128 loads (no div/mod, no scalar gather).
// v_wmma_f32_16x16x32_bf16 with fp32 accumulate. Selective scan is
// chunk-parallelized (exp(dt*A) products telescope to exp(A*sum dt)).
// out_proj applied after mean-pooling (linearity) -> ~17 GFLOP eliminated.
// ---------------------------------------------------------------------------

typedef __attribute__((ext_vector_type(16))) __bf16 v16bf;
typedef __attribute__((ext_vector_type(8)))  float  v8f;

#define BSZ      16
#define LSEQ     4096          // 64*64
#define BL       65536         // BSZ*LSEQ
#define NCHUNK   64
#define CLEN     64            // LSEQ / NCHUNK

__device__ __forceinline__ __bf16 f2bf(float f) { return (__bf16)f; }
__device__ __forceinline__ float  bf2f(__bf16 b) { return (float)b; }
__device__ __forceinline__ unsigned pack2bf(float lo, float hi) {
  union { __bf16 b[2]; unsigned u; } t;
  t.b[0] = (__bf16)lo; t.b[1] = (__bf16)hi; return t.u;
}
__device__ __forceinline__ float softplusf(float x) {
  return (x > 20.f) ? x : __logf(1.f + __expf(x));
}
__device__ __forceinline__ float siluf(float x) {
  return x / (1.f + __expf(-x));
}

// ----------------- A-matrix loaders: 8 consecutive k as packed bf16 --------
struct ABf16 {           // bf16 row-major: one B128 load, no conversion
  const __bf16* p; int ldk;
  __device__ uint4 load8(int m, int k) const {
    return *(const uint4*)(p + (size_t)m * ldk + k);
  }
};
struct APlain {          // fp32 row-major (tiny head GEMMs only)
  const float* p; int ldk;
  __device__ uint4 load8(int m, int k) const {
    const float4* q = (const float4*)(p + (size_t)m * ldk + k);
    float4 a = q[0], b = q[1];
    return make_uint4(pack2bf(a.x, a.y), pack2bf(a.z, a.w),
                      pack2bf(b.x, b.y), pack2bf(b.z, b.w));
  }
};
// im2col over channel-last bf16 h1 [16,128,128,32], tap-major K:
// k = tap*32 + c, tap = 3*(dy+1)+(dx+1). A k-tile is one tap, so the
// 8 fragment elements are 8 consecutive channels at one pixel: one B128
// load, zeroed via cndmask on the SAME-padding border. tap is scalar.
struct AIm2col {
  const __bf16* p;
  __device__ uint4 load8(int m, int k) const {
    int x = m & 127, y = (m >> 7) & 127, b = m >> 14;
    int tap = k >> 5, c0 = k & 31;               // tap uniform per k-tile
    int yy = y + tap / 3 - 1, xx = x + tap % 3 - 1;
    bool ok = ((unsigned)yy < 128u) & ((unsigned)xx < 128u);
    size_t idx = (((size_t)b * 128 + (ok ? yy : 0)) * 128 + (ok ? xx : 0)) * 32 + c0;
    uint4 v = *(const uint4*)(p + idx);
    if (!ok) v = make_uint4(0u, 0u, 0u, 0u);
    return v;
  }
};

// --------------------------- generic WMMA GEMM -----------------------------
// C[M,N] = act(A[M,K] * Wt[N,K]^T + bias), 16x16x32 bf16 WMMA, fp32 accum.
// Wt is pre-converted bf16 [N,ldw]. Block = 256 threads = 8 waves; wave ->
// one 16-row M tile x NT*16-col strip. K % 32 == 0, N == gridDim.y*NT*16.
template <typename AL, int NT>
__global__ __launch_bounds__(256)
void wmma_gemm(AL aload, const __bf16* __restrict__ Wt, const float* __restrict__ bias,
               float* __restrict__ outF, __bf16* __restrict__ outB,
               int Mtiles, int N, int K, int ldw, int act) {
  int wave = threadIdx.x >> 5;
  int lane = threadIdx.x & 31;
  int mtile = blockIdx.x * 8 + wave;
  if (mtile >= Mtiles) return;                   // wave-uniform: EXEC stays all-1s
  int nbase = blockIdx.y * (NT * 16);
  int m0 = mtile * 16;
  int rowA = m0 + (lane & 15);
  int kA = (lane & 16) ? 16 : 0;                 // A frag K-half select
  int kB = (lane & 16) ? 16 : 0;                 // B frag K-half select
  const __bf16* wrow = Wt + (size_t)(nbase + (lane & 15)) * ldw + kB;

  v8f acc[NT] = {};
  union VF { v16bf v; uint4 q[2]; } A, Bm;
  for (int k0 = 0; k0 < K; k0 += 32) {
    A.q[0] = aload.load8(rowA, k0 + kA);         // k = k0+kA .. +7
    A.q[1] = aload.load8(rowA, k0 + kA + 8);     // k = k0+kA+8 .. +15
    if (k0 + 32 < K)                             // global_prefetch of next W slab
      __builtin_prefetch(wrow + k0 + 32, 0, 0);
#pragma unroll
    for (int nt = 0; nt < NT; ++nt) {
      const uint4* w4 = (const uint4*)(wrow + (size_t)nt * 16 * ldw + k0);
      Bm.q[0] = w4[0];                           // raw bf16: two B128 loads
      Bm.q[1] = w4[1];
      acc[nt] = __builtin_amdgcn_wmma_f32_16x16x32_bf16(
          false, A.v, false, Bm.v, (short)0, acc[nt], false, false);
    }
  }
  // epilogue: lane holds C[m0 + r + 8*(lane>>4)][nbase + nt*16 + (lane&15)]
  int mrow = (lane >> 4) << 3;
#pragma unroll
  for (int nt = 0; nt < NT; ++nt) {
    int n = nbase + nt * 16 + (lane & 15);
    float bv = bias ? bias[n] : 0.f;
#pragma unroll
    for (int r = 0; r < 8; ++r) {
      float v = acc[nt][r] + bv;
      if (act == 1) v = softplusf(v);
      else if (act == 2) v = fmaxf(v, 0.f);
      size_t idx = (size_t)(m0 + mrow + r) * N + n;
      if (outB) outB[idx] = f2bf(v); else outF[idx] = v;
    }
  }
}

// ------------------------- fp32 -> bf16 weight copy ------------------------
__global__ __launch_bounds__(256)
void cvt_bf16_kernel(const float* __restrict__ s, __bf16* __restrict__ d, int n) {
  int t = blockIdx.x * 256 + threadIdx.x;
  if (t < n) d[t] = f2bf(s[t]);
}
// conv2 weights [64][32][3][3] -> bf16 tap-major [64][tap*32+c]
__global__ __launch_bounds__(256)
void cvt_conv2w_kernel(const float* __restrict__ s, __bf16* __restrict__ d) {
  int t = blockIdx.x * 256 + threadIdx.x;        // 64*288 = 18432
  if (t < 18432) {
    int n = t / 288, k = t - n * 288;
    int tap = k >> 5, c = k & 31;
    d[t] = f2bf(s[n * 288 + c * 9 + tap]);
  }
}

// -- conv1 (1->32, 3x3 SAME) + ReLU + maxpool2 -> bf16 channel-last h1 ------
__global__ __launch_bounds__(256)
void conv1_pool_kernel(const float* __restrict__ x, const float* __restrict__ w,
                       const float* __restrict__ bias, __bf16* __restrict__ h1) {
  int t = blockIdx.x * 256 + threadIdx.x;        // 16*128*128*32, c fastest
  int c = t & 31, ox = (t >> 5) & 127, oy = (t >> 12) & 127, b = t >> 19;
  float wv[9];
#pragma unroll
  for (int i = 0; i < 9; ++i) wv[i] = w[c * 9 + i];
  const float* xb = x + (size_t)b * 65536;
  float m = -1e30f;
#pragma unroll
  for (int dy = 0; dy < 2; ++dy)
#pragma unroll
    for (int dx = 0; dx < 2; ++dx) {
      int iy = oy * 2 + dy, ix = ox * 2 + dx;
      float s = 0.f;
#pragma unroll
      for (int ky = 0; ky < 3; ++ky) {
        int yy = iy + ky - 1;
        if ((unsigned)yy < 256u) {
#pragma unroll
          for (int kx = 0; kx < 3; ++kx) {
            int xx = ix + kx - 1;
            if ((unsigned)xx < 256u) s = fmaf(xb[yy * 256 + xx], wv[ky * 3 + kx], s);
          }
        }
      }
      m = fmaxf(m, s);
    }
  h1[t] = f2bf(fmaxf(m + bias[c], 0.f));         // [b][oy][ox][c]
}

// --------- relu+maxpool2 on conv2 output, emit bf16 [B,L,64] ---------------
__global__ __launch_bounds__(256)
void pool2_kernel(const float* __restrict__ h2pre, __bf16* __restrict__ a64) {
  int t = blockIdx.x * 256 + threadIdx.x;        // 16*64*64*64
  int c = t & 63, X = (t >> 6) & 63, Y = (t >> 12) & 63, b = t >> 18;
  size_t rb = (size_t)b * 128;
  float m =        h2pre[((rb + 2 * Y) * 128 + 2 * X) * 64 + c];
  m = fmaxf(m, h2pre[((rb + 2 * Y) * 128 + 2 * X + 1) * 64 + c]);
  m = fmaxf(m, h2pre[((rb + 2 * Y + 1) * 128 + 2 * X) * 64 + c]);
  m = fmaxf(m, h2pre[((rb + 2 * Y + 1) * 128 + 2 * X + 1) * 64 + c]);
  a64[t] = f2bf(fmaxf(m, 0.f));
}

// ----- depthwise causal conv1d (width 4) + SiLU -> xc fp32 + xcb bf16 ------
__global__ __launch_bounds__(256)
void dwconv_silu_kernel(const __bf16* __restrict__ xz, const float* __restrict__ w,
                        const float* __restrict__ bias, float* __restrict__ xc,
                        __bf16* __restrict__ xcb) {
  size_t t = (size_t)blockIdx.x * 256 + threadIdx.x; // 16*4096*512
  int d = (int)(t & 511);
  size_t bl = t >> 9;
  int l = (int)(bl & 4095);
  float acc = bias[d];
#pragma unroll
  for (int j = 0; j < 4; ++j) {
    int ll = l - 3 + j;
    if (ll >= 0) acc = fmaf(bf2f(xz[(bl - (size_t)(3 - j)) * 1024 + d]), w[d * 4 + j], acc);
  }
  float s = siluf(acc);
  xc[t] = s;
  xcb[t] = f2bf(s);
}

// ------------------------- chunked selective scan --------------------------
// thread -> (b, chunk, d);  d = dblk*32 + lane so a wave shares (b,l) and
// broadcasts the 48-wide x_dbl row via ds_bpermute (__shfl).
__global__ __launch_bounds__(256)
void scan_pass1(const float* __restrict__ xc, const float* __restrict__ xdbl,
                const float* __restrict__ dt_w, const float* __restrict__ dt_b,
                const float* __restrict__ A_log,
                float* __restrict__ hlocal, float* __restrict__ sdt_out) {
  int t = blockIdx.x * 256 + threadIdx.x;        // 16*64*16*32
  int lane = t & 31, dblk = (t >> 5) & 15, chunk = (t >> 9) & 63, b = t >> 15;
  int d = dblk * 32 + lane;
  float Arow[16], dtwr[16];
#pragma unroll
  for (int n = 0; n < 16; ++n) Arow[n] = -__expf(A_log[d * 16 + n]);
#pragma unroll
  for (int r = 0; r < 16; ++r) dtwr[r] = dt_w[d * 16 + r];
  float dtb = dt_b[d];
  float h[16];
#pragma unroll
  for (int n = 0; n < 16; ++n) h[n] = 0.f;
  float sdt = 0.f;
  size_t bl0 = (size_t)b * LSEQ + (size_t)chunk * CLEN;
  for (int s = 0; s < CLEN; ++s) {
    size_t bl = bl0 + s;
    float u  = xc[bl * 512 + d];
    float r0 = xdbl[bl * 48 + lane];             // lanes: [0..15]=dt_r, [16..31]=B
    float acc = dtb;
#pragma unroll
    for (int r = 0; r < 16; ++r) acc = fmaf(__shfl(r0, r, 32), dtwr[r], acc);
    float dtv = softplusf(acc);
    float du = dtv * u;
#pragma unroll
    for (int n = 0; n < 16; ++n) {
      float Bn = __shfl(r0, 16 + n, 32);
      h[n] = fmaf(h[n], __expf(dtv * Arow[n]), du * Bn);
    }
    sdt += dtv;
  }
  size_t o = ((size_t)(b * 512 + d)) * NCHUNK + chunk;
  sdt_out[o] = sdt;
#pragma unroll
  for (int n = 0; n < 16; ++n) hlocal[o * 16 + n] = h[n];
}

__global__ __launch_bounds__(256)
void scan_pass2(const float* __restrict__ A_log, const float* __restrict__ hlocal,
                const float* __restrict__ sdt, float* __restrict__ hstart) {
  int t = blockIdx.x * 256 + threadIdx.x;        // 8192 = b*512+d
  int d = t & 511;
  float Arow[16];
#pragma unroll
  for (int n = 0; n < 16; ++n) Arow[n] = -__expf(A_log[d * 16 + n]);
  float hs[16];
#pragma unroll
  for (int n = 0; n < 16; ++n) hs[n] = 0.f;
  size_t base = (size_t)t * NCHUNK;
  for (int c = 0; c < NCHUNK; ++c) {
    size_t o = base + c;
#pragma unroll
    for (int n = 0; n < 16; ++n) hstart[o * 16 + n] = hs[n];
    float S = sdt[o];
#pragma unroll
    for (int n = 0; n < 16; ++n)
      hs[n] = fmaf(hs[n], __expf(Arow[n] * S), hlocal[o * 16 + n]);
  }
}

__global__ __launch_bounds__(256)
void scan_pass3(const float* __restrict__ xc, const float* __restrict__ xdbl,
                const float* __restrict__ dt_w, const float* __restrict__ dt_b,
                const float* __restrict__ A_log, const float* __restrict__ D_vec,
                const __bf16* __restrict__ xz, const float* __restrict__ hstart,
                float* __restrict__ ychunk) {
  int t = blockIdx.x * 256 + threadIdx.x;
  int lane = t & 31, dblk = (t >> 5) & 15, chunk = (t >> 9) & 63, b = t >> 15;
  int d = dblk * 32 + lane;
  float Arow[16], dtwr[16];
#pragma unroll
  for (int n = 0; n < 16; ++n) Arow[n] = -__expf(A_log[d * 16 + n]);
#pragma unroll
  for (int r = 0; r < 16; ++r) dtwr[r] = dt_w[d * 16 + r];
  float dtb = dt_b[d], Dd = D_vec[d];
  size_t o = ((size_t)(b * 512 + d)) * NCHUNK + chunk;
  float h[16];
#pragma unroll
  for (int n = 0; n < 16; ++n) h[n] = hstart[o * 16 + n];
  float ysum = 0.f;
  size_t bl0 = (size_t)b * LSEQ + (size_t)chunk * CLEN;
  for (int s = 0; s < CLEN; ++s) {
    size_t bl = bl0 + s;
    float u  = xc[bl * 512 + d];
    float r0 = xdbl[bl * 48 + lane];
    float r1 = xdbl[bl * 48 + 32 + (lane & 15)]; // lanes dup C over halves
    float acc = dtb;
#pragma unroll
    for (int r = 0; r < 16; ++r) acc = fmaf(__shfl(r0, r, 32), dtwr[r], acc);
    float dtv = softplusf(acc);
    float du = dtv * u;
    float y = 0.f;
#pragma unroll
    for (int n = 0; n < 16; ++n) {
      float Bn = __shfl(r0, 16 + n, 32);
      float Cn = __shfl(r1, n, 32);
      h[n] = fmaf(h[n], __expf(dtv * Arow[n]), du * Bn);
      y = fmaf(h[n], Cn, y);
    }
    y = fmaf(u, Dd, y);
    ysum = fmaf(y, siluf(bf2f(xz[bl * 1024 + 512 + d])), ysum);
  }
  ychunk[o] = ysum;
}

__global__ __launch_bounds__(256)
void reduce_pool_kernel(const float* __restrict__ ychunk, float* __restrict__ ypool) {
  int t = blockIdx.x * 256 + threadIdx.x;        // 8192
  size_t base = (size_t)t * NCHUNK;
  float s = 0.f;
  for (int c = 0; c < NCHUNK; ++c) s += ychunk[base + c];
  ypool[t] = s * (1.f / (float)LSEQ);
}

// ------------------------------- classifier --------------------------------
__global__ void cls_kernel(const float* __restrict__ feat, const float* __restrict__ cw,
                           const float* __restrict__ cb, float* __restrict__ out) {
  int t = threadIdx.x;
  if (t < 32) {
    int b = t >> 1, j = t & 1;
    float s = cb[j];
    for (int k = 0; k < 128; ++k) s = fmaf(feat[b * 128 + k], cw[j * 128 + k], s);
    out[t] = s;
  }
}

// ---------------------------------------------------------------------------
extern "C" void kernel_launch(void* const* d_in, const int* in_sizes, int n_in,
                              void* d_out, int out_size, void* d_ws, size_t ws_size,
                              hipStream_t stream) {
  (void)in_sizes; (void)n_in; (void)out_size; (void)ws_size;
  const float* x         = (const float*)d_in[0];
  const float* conv1_w   = (const float*)d_in[1];
  const float* conv1_b   = (const float*)d_in[2];
  const float* conv2_w   = (const float*)d_in[3];
  const float* conv2_b   = (const float*)d_in[4];
  const float* pin_w     = (const float*)d_in[5];
  const float* pin_b     = (const float*)d_in[6];
  const float* in_proj_w = (const float*)d_in[7];
  const float* conv1d_w  = (const float*)d_in[8];
  const float* conv1d_b  = (const float*)d_in[9];
  const float* x_proj_w  = (const float*)d_in[10];
  const float* dt_w      = (const float*)d_in[11];
  const float* dt_b      = (const float*)d_in[12];
  const float* A_log     = (const float*)d_in[13];
  const float* D_vec     = (const float*)d_in[14];
  const float* out_proj_w= (const float*)d_in[15];
  const float* pout_w    = (const float*)d_in[16];
  const float* pout_b    = (const float*)d_in[17];
  const float* cls_w     = (const float*)d_in[18];
  const float* cls_b     = (const float*)d_in[19];
  float* out = (float*)d_out;

  char* ws = (char*)d_ws;
  // region A (conv-phase, later reused by scan temporaries)
  __bf16* h1b    = (__bf16*)(ws + 0);                 // 16,777,216 B [b][y][x][32]
  float*  h2pre  = (float*) (ws + 16777216ull);       // 67,108,864 B
  __bf16* a64b   = (__bf16*)(ws + 83886080ull);       //  8,388,608 B (ends 92,274,688)
  float*  hlocal = (float*) (ws + 0);                 // 33,554,432 B (alias)
  float*  hstart = (float*) (ws + 33554432ull);       // 33,554,432 B (alias)
  float*  sdt    = (float*) (ws + 67108864ull);       //  2,097,152 B (alias)
  float*  ychunk = (float*) (ws + 69206016ull);       //  2,097,152 B (alias)
  // persistent region
  const size_t P = 92274688ull;
  __bf16* seq    = (__bf16*)(ws + P);                 // 33,554,432 B
  __bf16* xz     = (__bf16*)(ws + P + 33554432ull);   // 134,217,728 B
  float*  xc     = (float*) (ws + P + 167772160ull);  // 134,217,728 B
  __bf16* xcb    = (__bf16*)(ws + P + 301989888ull);  // 67,108,864 B
  float*  xdbl   = (float*) (ws + P + 369098752ull);  // 12,582,912 B
  float*  ypool  = (float*) (ws + P + 381681664ull);  // 32,768 B
  float*  pooled = (float*) (ws + P + 381714432ull);  // 16,384 B
  float*  feat   = (float*) (ws + P + 381730816ull);  // 8,192 B
  char*   wb     = ws + P + 381739008ull;             // bf16 weight copies
  __bf16* w2b    = (__bf16*)(wb + 0);                 // 64*288   = 36,864 B (tap-major)
  __bf16* pinb   = (__bf16*)(wb + 36864);             // 256*64   = 32,768 B
  __bf16* inpb   = (__bf16*)(wb + 69632);             // 1024*256 = 524,288 B
  __bf16* xpb    = (__bf16*)(wb + 593920);            // 48*512   = 49,152 B
  __bf16* outpb  = (__bf16*)(wb + 643072);            // 256*512  = 262,144 B
  __bf16* poutb  = (__bf16*)(wb + 905216);            // 128*256  = 65,536 B

  // 0) one-shot weight conversion to bf16 (tiny; removes all per-tile cvt)
  cvt_conv2w_kernel<<<72, 256, 0, stream>>>(conv2_w, w2b);
  cvt_bf16_kernel<<<64, 256, 0, stream>>>(pin_w, pinb, 16384);
  cvt_bf16_kernel<<<1024, 256, 0, stream>>>(in_proj_w, inpb, 262144);
  cvt_bf16_kernel<<<96, 256, 0, stream>>>(x_proj_w, xpb, 24576);
  cvt_bf16_kernel<<<512, 256, 0, stream>>>(out_proj_w, outpb, 131072);
  cvt_bf16_kernel<<<128, 256, 0, stream>>>(pout_w, poutb, 32768);

  // 1) conv1 + relu + pool -> h1 bf16 channel-last [16,128,128,32]
  conv1_pool_kernel<<<32768, 256, 0, stream>>>(x, conv1_w, conv1_b, h1b);
  // 2) conv2 as implicit GEMM (WMMA): M=262144, N=64, K=288 -> h2pre [m][64]
  wmma_gemm<AIm2col, 4><<<dim3(2048, 1), 256, 0, stream>>>(
      AIm2col{h1b}, w2b, conv2_b, h2pre, (__bf16*)nullptr, 16384, 64, 288, 288, 0);
  // 3) relu + pool -> a64 bf16 [BL,64]
  pool2_kernel<<<16384, 256, 0, stream>>>(h2pre, a64b);
  // 4) pin: [BL,64] x [64,256] + bias -> seq bf16
  wmma_gemm<ABf16, 4><<<dim3(512, 4), 256, 0, stream>>>(
      ABf16{a64b, 64}, pinb, pin_b, (float*)nullptr, seq, 4096, 256, 64, 64, 0);
  // 5) in_proj: [BL,256] x [256,1024] -> xz bf16 (xm | z)
  wmma_gemm<ABf16, 4><<<dim3(512, 16), 256, 0, stream>>>(
      ABf16{seq, 256}, inpb, (const float*)nullptr, (float*)nullptr, xz,
      4096, 1024, 256, 256, 0);
  // 6) depthwise causal conv1d + SiLU -> xc fp32 (scan) + xcb bf16 (GEMM)
  dwconv_silu_kernel<<<131072, 256, 0, stream>>>(xz, conv1d_w, conv1d_b, xc, xcb);
  // 7) x_proj: [BL,512] x [512,48] -> xdbl fp32 (dt_r | B | C), NT=3 exact
  wmma_gemm<ABf16, 3><<<dim3(512, 1), 256, 0, stream>>>(
      ABf16{xcb, 512}, xpb, (const float*)nullptr, xdbl, (__bf16*)nullptr,
      4096, 48, 512, 512, 0);
  // 8-10) chunk-parallel selective scan (dt proj fused; silu(z)-gated pooling)
  scan_pass1<<<2048, 256, 0, stream>>>(xc, xdbl, dt_w, dt_b, A_log, hlocal, sdt);
  scan_pass2<<<32, 256, 0, stream>>>(A_log, hlocal, sdt, hstart);
  scan_pass3<<<2048, 256, 0, stream>>>(xc, xdbl, dt_w, dt_b, A_log, D_vec, xz,
                                       hstart, ychunk);
  // 11) reduce chunk partials, scale by 1/L -> ypool [16,512]
  reduce_pool_kernel<<<32, 256, 0, stream>>>(ychunk, ypool);
  // 12) out_proj on pooled vector (linearity with mean): [16,512]x[512,256]
  wmma_gemm<APlain, 4><<<dim3(1, 4), 256, 0, stream>>>(
      APlain{ypool, 512}, outpb, (const float*)nullptr, pooled, (__bf16*)nullptr,
      1, 256, 512, 512, 0);
  // 13) pout + bias + relu: [16,256]x[256,128]
  wmma_gemm<APlain, 4><<<dim3(1, 2), 256, 0, stream>>>(
      APlain{pooled, 256}, poutb, pout_b, feat, (__bf16*)nullptr,
      1, 128, 256, 256, 2);
  // 14) classifier head -> d_out [16,2]
  cls_kernel<<<1, 32, 0, stream>>>(feat, cls_w, cls_b, out);
}
